// PopcntLayer_14731737825610
// MI455X (gfx1250) — matrix operand order repaired and checked by reading, not compiled, not run on hardware
//
#include <hip/hip_runtime.h>

// ---------------------------------------------------------------------------
// PopcntLayer on MI455X (gfx1250):
//   out = sigmoid( gather(x,sel) *_p sigmoid(W) - bias )
// Reformulated as:  out = sigmoid( x @ W_dense - bias ),
//   W_dense[i,o] = sum_p sigmoid(W[o,p]) * [sel[o,p]==i]   (64 nnz / column)
// Pipeline: (1) x f32->bf16, (2) densify W into bf16 [o][k] layout via LDS
// atomic scatter, (3) double-buffered bf16 WMMA GEMM with async global->LDS.
// ---------------------------------------------------------------------------

typedef __attribute__((ext_vector_type(16))) __bf16 v16bf;
typedef __attribute__((ext_vector_type(8)))  float  v8f;
typedef __attribute__((ext_vector_type(4)))  int    v4i;

#define IW    4096   // INPUT_WIDTH  (= K of GEMM)
#define OW    4096   // OUTPUT_WIDTH (= N of GEMM)
#define PC    64     // POPCNT_WIDTH
#define BATCH 1024   // (= M of GEMM)

__device__ __forceinline__ float sigmoidf_(float x) {
    return 1.0f / (1.0f + __expf(-x));
}

__device__ __forceinline__ unsigned short f32_to_bf16(float f) {
    unsigned int u = __float_as_uint(f);
    unsigned int r = u + 0x7FFFu + ((u >> 16) & 1u);   // round-to-nearest-even
    return (unsigned short)(r >> 16);
}

// ---------------------------------------------------------------------------
// CDNA5 async global->LDS (ASYNCcnt path). Falls back to a sync copy if the
// builtin is absent on this toolchain.
// ---------------------------------------------------------------------------
#if __has_builtin(__builtin_amdgcn_global_load_async_to_lds_b128)
#define USE_ASYNC 1
#endif

typedef __attribute__((address_space(1))) v4i* gptr4;
typedef __attribute__((address_space(3))) v4i* lptr4;

__device__ __forceinline__ void copy16(const unsigned short* g, unsigned short* l) {
#ifdef USE_ASYNC
    __builtin_amdgcn_global_load_async_to_lds_b128(
        (gptr4)(size_t)g, (lptr4)(unsigned)(size_t)l, 0, 0);
#else
    *(uint4*)l = *(const uint4*)g;
#endif
}

__device__ __forceinline__ void wait_tile() {
#ifdef USE_ASYNC
#if __has_builtin(__builtin_amdgcn_s_wait_asynccnt)
    __builtin_amdgcn_s_wait_asynccnt(0);
#else
    asm volatile("s_wait_asynccnt 0" ::: "memory");
#endif
#endif
    __syncthreads();
}

// ---------------------------------------------------------------------------
// Kernel 1: x (f32, [B][IW]) -> bf16, 4 elements per thread
// ---------------------------------------------------------------------------
__global__ void __launch_bounds__(256)
x_to_bf16_kernel(const float* __restrict__ x, unsigned short* __restrict__ xb) {
    const size_t i = ((size_t)blockIdx.x * blockDim.x + threadIdx.x) * 4;
    const float4 v = *(const float4*)(x + i);
    union { unsigned short s[4]; uint2 u; } p;
    p.s[0] = f32_to_bf16(v.x);
    p.s[1] = f32_to_bf16(v.y);
    p.s[2] = f32_to_bf16(v.z);
    p.s[3] = f32_to_bf16(v.w);
    *(uint2*)(xb + i) = p.u;
}

// ---------------------------------------------------------------------------
// Kernel 2: densify.  Each block builds COLS columns of W_dense in LDS
// (f32, zeroed, ds_add_f32 scatter handles duplicate indices) and writes
// them out as bf16 in [o][k] (K-contiguous) layout for the GEMM B-matrix.
// ---------------------------------------------------------------------------
#define COLS 4
__global__ void __launch_bounds__(256)
densify_kernel(const int* __restrict__ sel, const float* __restrict__ w,
               unsigned short* __restrict__ Wt) {
    __shared__ __align__(16) float cols[COLS * IW];   // 64 KB
    const int tid   = threadIdx.x;
    const int oBase = blockIdx.x * COLS;

    for (int i = tid; i < COLS * IW / 4; i += 256)
        ((float4*)cols)[i] = make_float4(0.f, 0.f, 0.f, 0.f);
    __syncthreads();

    for (int e = tid; e < COLS * PC; e += 256) {
        const int c = e / PC, p = e % PC;
        const int o = oBase + c;
        const int k = sel[o * PC + p];
        atomicAdd(&cols[c * IW + k], sigmoidf_(w[o * PC + p]));
    }
    __syncthreads();

    for (int i = tid; i < COLS * IW / 2; i += 256) {
        const int c  = i / (IW / 2);
        const int kk = (i % (IW / 2)) * 2;
        const unsigned int pk = (unsigned int)f32_to_bf16(cols[c * IW + kk]) |
                                ((unsigned int)f32_to_bf16(cols[c * IW + kk + 1]) << 16);
        *(unsigned int*)(Wt + (size_t)(oBase + c) * IW + kk) = pk;
    }
}

// ---------------------------------------------------------------------------
// Kernel 3: bf16 WMMA GEMM. Block tile 128(M)x64(N), BK=64, 8 waves; each
// wave owns a 32x32 sub-tile (4 named accumulators, A/B fragment reuse).
// Double-buffered LDS with async global->LDS; ONE barrier per K-chunk.
// Steady-state loop body is branchless (last iteration peeled) so the
// accumulators stay pinned in place across iterations.
// LDS rows padded to 72 elements (36 dwords) for conflict-free ds_load_b128.
// ---------------------------------------------------------------------------
#define BM 128
#define BN 64
#define BK 64
#define LDA 72
#define LDB 72

#define WMMA_BF16(A, B, C) \
    __builtin_amdgcn_wmma_f32_16x16x32_bf16(false, (A), false, (B), (short)0, (C), false, false)

__device__ __forceinline__ void issue_tile(const unsigned short* __restrict__ xb,
                                           const unsigned short* __restrict__ Wt,
                                           unsigned short* Ab, unsigned short* Bb,
                                           int mBlock, int nBlock, int kc, int tid) {
#pragma unroll
    for (int t = 0; t < 4; ++t) {                       // A: 128 rows x 8 chunks
        const int i   = tid + t * 256;
        const int row = i >> 3;
        const int seg = (i & 7) * 8;
        copy16(xb + (size_t)(mBlock + row) * IW + kc + seg, Ab + row * LDA + seg);
    }
#pragma unroll
    for (int t = 0; t < 2; ++t) {                       // B: 64 rows x 8 chunks
        const int i   = tid + t * 256;
        const int row = i >> 3;
        const int seg = (i & 7) * 8;
        copy16(Wt + (size_t)(nBlock + row) * IW + kc + seg, Bb + row * LDB + seg);
    }
}

__device__ __forceinline__ void compute_chunk(const unsigned short* __restrict__ Ab,
                                              const unsigned short* __restrict__ Bb,
                                              int wm, int wn, int am, int ah, int bn, int bh,
                                              v8f& acc00, v8f& acc01, v8f& acc10, v8f& acc11) {
#pragma unroll
    for (int ks = 0; ks < BK; ks += 32) {
        union { uint4 u[2]; v16bf v; } a0, a1, b0, b1;
        const unsigned short* ap0 = Ab + (wm * 32 + am) * LDA + ks + ah;
        const unsigned short* ap1 = ap0 + 16 * LDA;
        a0.u[0] = *(const uint4*)(ap0);
        a0.u[1] = *(const uint4*)(ap0 + 16);
        a1.u[0] = *(const uint4*)(ap1);
        a1.u[1] = *(const uint4*)(ap1 + 16);
        const unsigned short* bp0 = Bb + (wn * 32 + bn) * LDB + ks + bh;
        const unsigned short* bp1 = bp0 + 16 * LDB;
        b0.u[0] = *(const uint4*)(bp0);
        b0.u[1] = *(const uint4*)(bp0 + 16);
        b1.u[0] = *(const uint4*)(bp1);
        b1.u[1] = *(const uint4*)(bp1 + 16);

        acc00 = WMMA_BF16(a0.v, b0.v, acc00);
        acc01 = WMMA_BF16(a0.v, b1.v, acc01);
        acc10 = WMMA_BF16(a1.v, b0.v, acc10);
        acc11 = WMMA_BF16(a1.v, b1.v, acc11);
    }
}

__global__ void __launch_bounds__(256)
gemm_kernel(const unsigned short* __restrict__ xb,   // [BATCH][IW] bf16
            const unsigned short* __restrict__ Wt,   // [OW][IW]    bf16
            const float* __restrict__ bias,
            float* __restrict__ out) {
    __shared__ __align__(16) unsigned short As[2][BM * LDA];  // 2 x 18 KB
    __shared__ __align__(16) unsigned short Bs[2][BN * LDB];  // 2 x  9 KB

    const int tid  = threadIdx.x;
    const int lane = tid & 31;
    const int wave = tid >> 5;
    const int wm   = wave >> 1;      // 0..3 : 32-row M sub-tile
    const int wn   = wave & 1;       // 0..1 : 32-col N sub-tile
    const int mBlock = blockIdx.y * BM;
    const int nBlock = blockIdx.x * BN;

    v8f acc00 = {}, acc01 = {}, acc10 = {}, acc11 = {};

    // CDNA5 wave32 fragment coordinates
    const int am = lane & 15;          // A: M = lane%16
    const int ah = (lane >> 4) * 8;    // A: K-half offset (0 or 8)
    const int bn = lane & 15;          // B: N = lane%16
    const int bh = (lane >> 4) * 8;    // B: K-half offset (0 or 8)

    const int NK = IW / BK;            // 64 K-chunks
    issue_tile(xb, Wt, As[0], Bs[0], mBlock, nBlock, 0, tid);

    // Steady state: branchless body (wait -> issue next -> compute current).
    for (int it = 0; it < NK - 1; ++it) {
        wait_tile();
        issue_tile(xb, Wt, As[(it + 1) & 1], Bs[(it + 1) & 1],
                   mBlock, nBlock, (it + 1) * BK, tid);
        compute_chunk(As[it & 1], Bs[it & 1], wm, wn, am, ah, bn, bh,
                      acc00, acc01, acc10, acc11);
    }
    // Peeled tail.
    wait_tile();
    compute_chunk(As[(NK - 1) & 1], Bs[(NK - 1) & 1], wm, wn, am, ah, bn, bh,
                  acc00, acc01, acc10, acc11);

    // Epilogue. C/D layout: VGPR j -> M = j + 8*(lane>=16), N = lane%16.
    const int ln = lane & 15;
    const int lh = (lane >> 4) * 8;
    const int r0   = mBlock + wm * 32 + lh;          // mi = 0
    const int r1   = r0 + 16;                        // mi = 1
    const int col0 = nBlock + wn * 32 + ln;          // ni = 0
    const int col1 = col0 + 16;                      // ni = 1
    const float bv0 = bias[col0];
    const float bv1 = bias[col1];
#pragma unroll
    for (int j = 0; j < 8; ++j) {
        out[(size_t)(r0 + j) * OW + col0] = sigmoidf_(acc00[j] - bv0);
        out[(size_t)(r0 + j) * OW + col1] = sigmoidf_(acc01[j] - bv1);
        out[(size_t)(r1 + j) * OW + col0] = sigmoidf_(acc10[j] - bv0);
        out[(size_t)(r1 + j) * OW + col1] = sigmoidf_(acc11[j] - bv1);
    }
}

// ---------------------------------------------------------------------------
// Fallback (only if workspace is too small for the 40 MB densified path):
// LDS-staged direct gather, one block per batch row.
// ---------------------------------------------------------------------------
__global__ void __launch_bounds__(256)
fallback_kernel(const float* __restrict__ x, const int* __restrict__ sel,
                const float* __restrict__ w, const float* __restrict__ bias,
                float* __restrict__ out) {
    __shared__ float xr[IW];
    const int b = blockIdx.x;
    for (int i = threadIdx.x; i < IW; i += blockDim.x)
        xr[i] = x[(size_t)b * IW + i];
    __syncthreads();
    for (int o = threadIdx.x; o < OW; o += blockDim.x) {
        float s = 0.f;
#pragma unroll 4
        for (int p = 0; p < PC; ++p)
            s += xr[sel[o * PC + p]] * sigmoidf_(w[o * PC + p]);
        out[(size_t)b * OW + o] = sigmoidf_(s - bias[o]);
    }
}

// ---------------------------------------------------------------------------
extern "C" void kernel_launch(void* const* d_in, const int* in_sizes, int n_in,
                              void* d_out, int out_size, void* d_ws, size_t ws_size,
                              hipStream_t stream) {
    const float* x    = (const float*)d_in[0];
    const int*   sel  = (const int*)d_in[1];
    const float* w    = (const float*)d_in[2];
    const float* bias = (const float*)d_in[3];
    float*       out  = (float*)d_out;

    const size_t xb_bytes = (size_t)BATCH * IW * sizeof(unsigned short); //  8 MB
    const size_t wt_bytes = (size_t)OW * IW * sizeof(unsigned short);    // 32 MB

    if (ws_size >= xb_bytes + wt_bytes) {
        unsigned short* xb = (unsigned short*)d_ws;
        unsigned short* Wt = (unsigned short*)((char*)d_ws + xb_bytes);

        x_to_bf16_kernel<<<(BATCH * IW) / (256 * 4), 256, 0, stream>>>(x, xb);
        densify_kernel<<<OW / COLS, 256, 0, stream>>>(sel, w, Wt);
        gemm_kernel<<<dim3(OW / BN, BATCH / BM), 256, 0, stream>>>(xb, Wt, bias, out);
    } else {
        fallback_kernel<<<BATCH, 256, 0, stream>>>(x, sel, w, bias, out);
    }
}
